// TransformerBlock_4234837753859
// MI455X (gfx1250) — matrix operand order repaired and checked by reading, not compiled, not run on hardware
//
#include <hip/hip_runtime.h>
#include <hip/hip_bf16.h>
#include <math.h>

// ---------------------------------------------------------------------------
// Types for CDNA5 WMMA (wave32): V_WMMA_F32_16X16X32_BF16
// ---------------------------------------------------------------------------
typedef __bf16 bf16;
typedef __attribute__((ext_vector_type(16))) __bf16 v16bf;
typedef __attribute__((ext_vector_type(8)))  __bf16 v8bf;
typedef __attribute__((ext_vector_type(8)))  float   v8f;
typedef __attribute__((ext_vector_type(4)))  unsigned u32x4;
typedef __attribute__((ext_vector_type(8)))  unsigned u32x8;

#define EMBED  768
#define HEADS  12
#define HDIM   64
#define HIDDEN 3072
#define BATCH  4
#define SEQ    2048
#define MTOK   (BATCH*SEQ)   /* 8192 token rows */

union Frag16 { v16bf v; v8bf h[2]; };

__device__ __forceinline__ bf16 to_bf16(float f) { return (bf16)f; }

// A-fragment (16x32 bf16, MxK). ISA 7.12.2: lanes 0-15 hold row M=lane,
// K {0..7,16..23}; lanes 16-31 hold row M=lane-16, K {8..15,24..31}.
__device__ __forceinline__ v16bf frag_a(const bf16* base, int stride, int row, int kb) {
  Frag16 f;
  f.h[0] = *(const v8bf*)(base + row*stride + kb);
  f.h[1] = *(const v8bf*)(base + row*stride + kb + 16);
  return f.v;
}
// B-fragment (32x16 bf16, KxN) from a K-contiguous LDS tile [n][k]:
// lanes 0-15 hold col N=lane, K {0..7,16..23}; lanes 16-31 hold K {8..15,24..31}.
__device__ __forceinline__ v16bf frag_b(const bf16* base, int stride, int col, int kb) {
  Frag16 f;
  f.h[0] = *(const v8bf*)(base + col*stride + kb);
  f.h[1] = *(const v8bf*)(base + col*stride + kb + 16);
  return f.v;
}

__device__ __forceinline__ v8f wmma_bf16(v16bf a, v16bf b, v8f c) {
  return __builtin_amdgcn_wmma_f32_16x16x32_bf16(false, a, false, b, (short)0, c,
                                                 false, false);
}

// ---------------------------------------------------------------------------
// Pure-VALU row reductions via DPP16 (no ds_bpermute, no DScnt waits).
// ---------------------------------------------------------------------------
template <int CTRL>
__device__ __forceinline__ float dpp_xf(float x) {
  return __builtin_bit_cast(float,
      __builtin_amdgcn_mov_dpp(__builtin_bit_cast(int, x), CTRL, 0xF, 0xF, true));
}
__device__ __forceinline__ float row16_max(float x) {
  x = fmaxf(x, dpp_xf<0xB1>(x));   // quad_perm(1,0,3,2) : xor 1
  x = fmaxf(x, dpp_xf<0x4E>(x));   // quad_perm(2,3,0,1) : xor 2
  x = fmaxf(x, dpp_xf<0x141>(x));  // row_half_mirror
  x = fmaxf(x, dpp_xf<0x140>(x));  // row_mirror
  return x;
}
__device__ __forceinline__ float row16_sum(float x) {
  x += dpp_xf<0xB1>(x);
  x += dpp_xf<0x4E>(x);
  x += dpp_xf<0x141>(x);
  x += dpp_xf<0x140>(x);
  return x;
}

// ---------------------------------------------------------------------------
// Tensor Data Mover: 2D tile load global->LDS via inline asm (portable across
// toolchains per CDNA5_HIP.md). D# groups per CDNA5 ISA.
// Padding: after every 32 DWORDs (one 64-elem bf16 row) insert 4 DWORDs
// (8 bf16) -> LDS row stride 72 elements, matching the frag layout below.
// ---------------------------------------------------------------------------
__device__ __forceinline__ unsigned lds_offset(const void* p) {
  return (unsigned)(unsigned long long)
      (__attribute__((address_space(3))) const void*)p;
}

__device__ __forceinline__ void tdm_load_2d_bf16(
    unsigned lds_addr, unsigned long long gaddr,
    unsigned tensor_d0, unsigned tensor_d1,
    unsigned tile_d0, unsigned tile_d1, unsigned long long stride0) {
  u32x4 g0;
  g0[0] = 1u;                                            // count=1 (valid, user)
  g0[1] = lds_addr;                                      // LDS byte address
  g0[2] = (unsigned)(gaddr & 0xFFFFFFFFull);             // global_addr[31:0]
  g0[3] = (unsigned)((gaddr >> 32) & 0x01FFFFFFull)      // global_addr[56:32]
        | (2u << 30);                                    // type = 2 ("image")
  u32x8 g1;
  g1[0] = (1u << 16)      // data_size = 1 -> 2 bytes
        | (1u << 20)      // pad_enable
        | (4u << 22)      // pad_interval code 4 -> 32 DWORDs
        | (3u << 25);     // pad_amount  code 3 -> 4 DWORDs
  g1[1] = (tensor_d0 & 0xFFFFu) << 16;                   // tensor_dim0[15:0]
  g1[2] = (tensor_d0 >> 16) | ((tensor_d1 & 0xFFFFu) << 16);
  g1[3] = (tensor_d1 >> 16) | (tile_d0 << 16);
  g1[4] = tile_d1;                                       // tile_dim2 = 0
  g1[5] = (unsigned)(stride0 & 0xFFFFFFFFull);           // dim0 stride (elems)
  g1[6] = (unsigned)((stride0 >> 32) & 0xFFFFull);       // dim1 stride = 0
  g1[7] = 0u;
  u32x4 gz = {0u, 0u, 0u, 0u};                           // groups 2/3: unused (2D)
  asm volatile("tensor_load_to_lds %0, %1, %2, %3"
               :: "s"(g0), "s"(g1), "s"(gz), "s"(gz)
               : "memory");
}

// ---------------------------------------------------------------------------
// One-time fp32 [K,N] -> bf16 [N,K] transpose-convert.
// ---------------------------------------------------------------------------
__global__ void cvt_transpose_bf16(const float* __restrict__ in, bf16* __restrict__ out,
                                   int K, int N) {
  int idx = blockIdx.x * blockDim.x + threadIdx.x;
  if (idx < K * N) {
    int k = idx / N, n = idx - k * N;        // reads coalesced along n
    out[(size_t)n * K + k] = to_bf16(in[idx]);
  }
}

// ---------------------------------------------------------------------------
// LayerNorm over C=768, one row per 256-thread block, bf16 output
// ---------------------------------------------------------------------------
__global__ __launch_bounds__(256)
void layernorm_bf16(const float* __restrict__ x, const float* __restrict__ g,
                    const float* __restrict__ b, bf16* __restrict__ out) {
  const int row = blockIdx.x;
  const float* xr = x + (size_t)row * EMBED;
  float v[3], s = 0.f, s2 = 0.f;
#pragma unroll
  for (int i = 0; i < 3; i++) {
    int idx = threadIdx.x + i * 256;
    float t = xr[idx];
    v[i] = t; s += t; s2 += t * t;
  }
  s  = row16_sum(s)  + __shfl_xor(row16_sum(s),  16, 32);
  s2 = row16_sum(s2) + __shfl_xor(row16_sum(s2), 16, 32);
  __shared__ float red[16];
  int wave = threadIdx.x >> 5, lane = threadIdx.x & 31;
  if (lane == 0) { red[wave] = s; red[8 + wave] = s2; }
  __syncthreads();
  float ts = 0.f, ts2 = 0.f;
#pragma unroll
  for (int w = 0; w < 8; w++) { ts += red[w]; ts2 += red[8 + w]; }
  float mu   = ts  * (1.0f / EMBED);
  float var  = ts2 * (1.0f / EMBED) - mu * mu;
  float rstd = rsqrtf(var + 1e-5f);
  bf16* orow = out + (size_t)row * EMBED;
#pragma unroll
  for (int i = 0; i < 3; i++) {
    int idx = threadIdx.x + i * 256;
    orow[idx] = to_bf16((v[i] - mu) * rstd * g[idx] + b[idx]);
  }
}

// ---------------------------------------------------------------------------
// bf16 GEMM: C[M,N] = A[M,K] * Wt[N,K]^T (+bias, mode-specific epilogue)
// Block tile 128x128, 8 waves (2x4) of 64x32, K-step 32, double-buffered LDS.
// MODE 0: bf16 QKV split-store; Q pre-scaled by 1/sqrt(64); V stored [B,H,D,N]
// MODE 1: fp32 out = acc + bias + residual
// MODE 2: bf16 out = gelu(acc + bias)
// ---------------------------------------------------------------------------
#define BM 128
#define BN 128
#define BK 32
#define KP 40   /* LDS K-stride (pad 8): 80B rows keep 16B alignment */

template <int MODE>
__global__ __launch_bounds__(256)
void gemm_bf16(const bf16* __restrict__ A, const bf16* __restrict__ Wt,
               const float* __restrict__ bias, const float* __restrict__ res,
               float* __restrict__ outF, bf16* __restrict__ outB,
               bf16* __restrict__ outQ, bf16* __restrict__ outK,
               bf16* __restrict__ outV, int M, int N, int K) {
  __shared__ bf16 sA[2][BM * KP];
  __shared__ bf16 sB[2][BN * KP];

  const int tid  = threadIdx.x;
  const int bn   = blockIdx.x * BN, bm = blockIdx.y * BM;
  const int wave = tid >> 5, lane = tid & 31;
  const int half = lane >> 4, l15 = lane & 15;
  const int wm = (wave >> 2) * 64, wn = (wave & 3) * 32;
  const int ka = half ? 8 : 0;
  const int kb = half ? 8 : 0;

  v8f acc[4][2] = {};
  const int nk = K / BK;

  int trow[2], tpart[2];
#pragma unroll
  for (int s = 0; s < 2; s++) {
    int seg = tid + s * 256;            // 512 x 16B segments per tile
    trow[s] = seg >> 2; tpart[s] = seg & 3;
  }

  v8bf ra[2], rb[2];
  auto g_load = [&](int kt, int s) {
    ra[s] = *(const v8bf*)(A  + (size_t)(bm + trow[s]) * K + kt * BK + tpart[s] * 8);
    rb[s] = *(const v8bf*)(Wt + (size_t)(bn + trow[s]) * K + kt * BK + tpart[s] * 8);
  };
  auto l_store = [&](int buf, int s) {
    *(v8bf*)(&sA[buf][trow[s] * KP + tpart[s] * 8]) = ra[s];
    *(v8bf*)(&sB[buf][trow[s] * KP + tpart[s] * 8]) = rb[s];
  };

  g_load(0, 0); g_load(0, 1);
  l_store(0, 0); l_store(0, 1);
  __syncthreads();

  for (int kt = 0; kt < nk; ++kt) {
    const int cur = kt & 1;
    const bool more = (kt + 1 < nk);
    if (more) {
      __builtin_prefetch(A  + (size_t)(bm + trow[0]) * K + (kt + 1) * BK, 0, 0);
      __builtin_prefetch(Wt + (size_t)(bn + trow[0]) * K + (kt + 1) * BK, 0, 0);
      g_load(kt + 1, 0); g_load(kt + 1, 1);
    }
    const bf16* Ab = sA[cur];
    const bf16* Bb = sB[cur];
    v16bf bfrag[2];
#pragma unroll
    for (int nt = 0; nt < 2; nt++)
      bfrag[nt] = frag_b(Bb, KP, wn + nt * 16 + l15, kb);
#pragma unroll
    for (int mt = 0; mt < 4; mt++) {
      v16bf af = frag_a(Ab, KP, wm + mt * 16 + l15, ka);
#pragma unroll
      for (int nt = 0; nt < 2; nt++)
        acc[mt][nt] = wmma_bf16(af, bfrag[nt], acc[mt][nt]);
    }
    if (more) { l_store(cur ^ 1, 0); l_store(cur ^ 1, 1); }
    __syncthreads();
  }

  // Epilogue. C/D layout: VGPR r -> row (r + half*8), lane&15 -> col.
#pragma unroll
  for (int mt = 0; mt < 4; mt++) {
#pragma unroll
    for (int nt = 0; nt < 2; nt++) {
#pragma unroll
      for (int r = 0; r < 8; r++) {
        int row = bm + wm + mt * 16 + r + half * 8;
        int col = bn + wn + nt * 16 + l15;
        float val = acc[mt][nt][r] + bias[col];
        if (MODE == 0) {
          int s3 = col / EMBED, rem = col - s3 * EMBED;
          int hh = rem >> 6, dd = rem & 63;
          int bb = row >> 11, nn = row & (SEQ - 1);
          if (s3 == 0) {        // Q: fold softmax scale (exact power of two)
            size_t dst = ((size_t)((bb * HEADS + hh) * SEQ + nn)) * HDIM + dd;
            outQ[dst] = to_bf16(val * 0.125f);
          } else if (s3 == 1) { // K: [B,H,N,D]
            size_t dst = ((size_t)((bb * HEADS + hh) * SEQ + nn)) * HDIM + dd;
            outK[dst] = to_bf16(val);
          } else {              // V: transposed [B,H,D,N]
            size_t dst = ((size_t)((bb * HEADS + hh) * HDIM + dd)) * SEQ + nn;
            outV[dst] = to_bf16(val);
          }
        } else if (MODE == 1) {
          size_t idx = (size_t)row * N + col;
          outF[idx] = val + res[idx];
        } else {
          float gl = 0.5f * val * (1.0f + erff(val * 0.70710678118654752f));
          outB[(size_t)row * N + col] = to_bf16(gl);
        }
      }
    }
  }
}

// ---------------------------------------------------------------------------
// Flash attention: one (b, h, 128-query tile) per 256-thread block (8 waves,
// 16 query rows each). 64-key tiles staged by the Tensor Data Mover (wave 0
// issues two 2D tile DMAs, waits TENSORcnt, barrier publishes to all waves).
// Online softmax with DPP reductions; WMMA for Q*K^T and P*V.
// ---------------------------------------------------------------------------
#define DP 72  /* LDS stride: 64 + 8 pad elements (matches TDM pad fields) */

__global__ __launch_bounds__(256)
void flash_attn(const bf16* __restrict__ Q, const bf16* __restrict__ Kb,
                const bf16* __restrict__ Vt_g, bf16* __restrict__ O) {
  __shared__ bf16 Ks[64 * DP];        // [key][d]  (B-operand for Q*K^T)
  __shared__ bf16 Vt[64 * DP];        // [d][key]  (B-operand for P*V)
  __shared__ bf16 Ps[8 * 16 * DP];    // per-wave P staging [row][key]

  const int tid = threadIdx.x;
  const int wave = tid >> 5, lane = tid & 31;
  const int half = lane >> 4, l15 = lane & 15;
  const int b = blockIdx.z, h = blockIdx.y;
  const int qb = blockIdx.x * 128;
  const size_t bh = ((size_t)b * HEADS + h) * SEQ * HDIM;
  const bf16* Qh = Q + bh;
  const bf16* Kh = Kb + bh;
  const bf16* Vh = Vt_g + bh;          // [d][n] for this (b,h)

  const unsigned ks_lds = lds_offset(&Ks[0]);
  const unsigned vt_lds = lds_offset(&Vt[0]);

  // Q fragments (pre-scaled bf16), kept in registers for the whole kernel.
  v16bf aq[2];
  {
    const bf16* qrow = Qh + (size_t)(qb + wave * 16 + l15) * HDIM;
    const int kq = half ? 8 : 0;
#pragma unroll
    for (int c = 0; c < 2; c++) {
      Frag16 f;
      f.h[0] = *(const v8bf*)(qrow + c * 32 + kq);
      f.h[1] = *(const v8bf*)(qrow + c * 32 + kq + 16);
      aq[c] = f.v;
    }
  }

  float m_r[8], l_r[8];
  v8f o_acc[4] = {};
#pragma unroll
  for (int r = 0; r < 8; r++) { m_r[r] = -1e30f; l_r[r] = 0.f; }

  bf16* pw = Ps + wave * 16 * DP;
  const int kaP = half ? 8 : 0;
  const int kbB = half ? 8 : 0;

  for (int j = 0; j < SEQ / 64; ++j) {
    const int key0 = j * 64;
    // --- TDM: stage K (64 keys x 64 d) and V^T (64 d x 64 keys) tiles ---
    if (wave == 0) {
      tdm_load_2d_bf16(ks_lds,
                       (unsigned long long)(const void*)(Kh + (size_t)key0 * HDIM),
                       /*tensor_d0=*/HDIM, /*tensor_d1=*/SEQ,
                       /*tile_d0=*/64, /*tile_d1=*/64, /*stride0=*/HDIM);
      tdm_load_2d_bf16(vt_lds,
                       (unsigned long long)(const void*)(Vh + key0),
                       /*tensor_d0=*/SEQ, /*tensor_d1=*/HDIM,
                       /*tile_d0=*/64, /*tile_d1=*/64, /*stride0=*/SEQ);
      asm volatile("s_wait_tensorcnt 0x0" ::: "memory");
    }
    __syncthreads();

    // --- S = Q K^T (16x64 per wave) ---
    v8f s_acc[4] = {};
#pragma unroll
    for (int c = 0; c < 2; c++) {
#pragma unroll
      for (int nt = 0; nt < 4; nt++) {
        v16bf bk = frag_b(Ks, DP, nt * 16 + l15, c * 32 + kbB);
        s_acc[nt] = wmma_bf16(aq[c], bk, s_acc[nt]);
      }
    }

    // --- online softmax; rows live in 16-lane halves -> DPP reductions ---
#pragma unroll
    for (int r = 0; r < 8; r++) {
      float mx = row16_max(fmaxf(fmaxf(s_acc[0][r], s_acc[1][r]),
                                 fmaxf(s_acc[2][r], s_acc[3][r])));
      float mnew  = fmaxf(m_r[r], mx);
      float alpha = __expf(m_r[r] - mnew);
      m_r[r] = mnew;
      float rs = 0.f;
#pragma unroll
      for (int nt = 0; nt < 4; nt++) {
        float p = __expf(s_acc[nt][r] - mnew);
        s_acc[nt][r] = p; rs += p;
      }
      l_r[r] = l_r[r] * alpha + row16_sum(rs);
#pragma unroll
      for (int nt = 0; nt < 4; nt++) o_acc[nt][r] *= alpha;
    }

    // --- stage P (bf16) in this wave's private LDS region ---
#pragma unroll
    for (int nt = 0; nt < 4; nt++)
#pragma unroll
      for (int r = 0; r < 8; r++)
        pw[(r + half * 8) * DP + nt * 16 + l15] = to_bf16(s_acc[nt][r]);
    // Same-wave DS ops are in-order; we only need our own stores retired.
    asm volatile("s_wait_dscnt 0x0" ::: "memory");

    // --- O += P V ---
#pragma unroll
    for (int c = 0; c < 2; c++) {
      Frag16 fa;
      const bf16* pr = pw + l15 * DP + c * 32 + kaP;
      fa.h[0] = *(const v8bf*)pr;
      fa.h[1] = *(const v8bf*)(pr + 16);
#pragma unroll
      for (int nt = 0; nt < 4; nt++) {
        v16bf bv = frag_b(Vt, DP, nt * 16 + l15, c * 32 + kbB);
        o_acc[nt] = wmma_bf16(fa.v, bv, o_acc[nt]);
      }
    }
    __syncthreads();   // protect Ks/Vt before next tile's DMA overwrites them
  }

  // --- normalize and store to [B,N,C] with col = h*64 + d ---
#pragma unroll
  for (int r = 0; r < 8; r++) {
    float inv = 1.0f / l_r[r];
    int row = qb + wave * 16 + r + half * 8;
#pragma unroll
    for (int nt = 0; nt < 4; nt++) {
      int d = nt * 16 + l15;
      O[((size_t)b * SEQ + row) * EMBED + h * HDIM + d] = to_bf16(o_acc[nt][r] * inv);
    }
  }
}

// ---------------------------------------------------------------------------
// Orchestration
// ---------------------------------------------------------------------------
extern "C" void kernel_launch(void* const* d_in, const int* in_sizes, int n_in,
                              void* d_out, int out_size, void* d_ws, size_t ws_size,
                              hipStream_t stream) {
  (void)in_sizes; (void)n_in; (void)out_size; (void)ws_size;
  const float* x      = (const float*)d_in[0];
  const float* ln1_g  = (const float*)d_in[1];
  const float* ln1_b  = (const float*)d_in[2];
  const float* qkv_w  = (const float*)d_in[3];
  const float* qkv_b  = (const float*)d_in[4];
  const float* proj_w = (const float*)d_in[5];
  const float* proj_b = (const float*)d_in[6];
  const float* ln2_g  = (const float*)d_in[7];
  const float* ln2_b  = (const float*)d_in[8];
  const float* fc1_w  = (const float*)d_in[9];
  const float* fc1_b  = (const float*)d_in[10];
  const float* fc2_w  = (const float*)d_in[11];
  const float* fc2_b  = (const float*)d_in[12];

  char* ws = (char*)d_ws;
  size_t off = 0;
  auto alloc = [&](size_t bytes) -> void* {
    void* p = ws + off;
    off += (bytes + 255) & ~(size_t)255;
    return p;
  };
  bf16*  wq  = (bf16*)alloc((size_t)EMBED * 3 * EMBED * 2);   // [3C, C]
  bf16*  wp  = (bf16*)alloc((size_t)EMBED * EMBED * 2);       // [C, C]
  bf16*  w1  = (bf16*)alloc((size_t)EMBED * HIDDEN * 2);      // [H, C]
  bf16*  w2  = (bf16*)alloc((size_t)HIDDEN * EMBED * 2);      // [C, H]
  bf16*  xn  = (bf16*)alloc((size_t)MTOK * EMBED * 2);
  bf16*  qb_ = (bf16*)alloc((size_t)MTOK * EMBED * 2);
  bf16*  kb_ = (bf16*)alloc((size_t)MTOK * EMBED * 2);
  bf16*  vb_ = (bf16*)alloc((size_t)MTOK * EMBED * 2);
  bf16*  ao  = (bf16*)alloc((size_t)MTOK * EMBED * 2);
  float* x1  = (float*)alloc((size_t)MTOK * EMBED * 4);
  // q/k/v/ao region == MTOK*HIDDEN*2 bytes exactly: reuse for MLP hidden h.
  bf16* hbuf = qb_;

  // 1) weights -> bf16, transposed to [N,K]
  { int n = EMBED * 3 * EMBED; cvt_transpose_bf16<<<(n + 255) / 256, 256, 0, stream>>>(qkv_w, wq, EMBED, 3 * EMBED); }
  { int n = EMBED * EMBED;     cvt_transpose_bf16<<<(n + 255) / 256, 256, 0, stream>>>(proj_w, wp, EMBED, EMBED); }
  { int n = EMBED * HIDDEN;    cvt_transpose_bf16<<<(n + 255) / 256, 256, 0, stream>>>(fc1_w, w1, EMBED, HIDDEN); }
  { int n = HIDDEN * EMBED;    cvt_transpose_bf16<<<(n + 255) / 256, 256, 0, stream>>>(fc2_w, w2, HIDDEN, EMBED); }

  // 2) LN1 -> bf16
  layernorm_bf16<<<MTOK, 256, 0, stream>>>(x, ln1_g, ln1_b, xn);

  // 3) QKV GEMM (split-store; Q pre-scaled; V transposed)
  gemm_bf16<0><<<dim3(3 * EMBED / BN, MTOK / BM), 256, 0, stream>>>(
      xn, wq, qkv_b, nullptr, nullptr, nullptr, qb_, kb_, vb_, MTOK, 3 * EMBED, EMBED);

  // 4) flash attention -> ao [B,N,C]
  flash_attn<<<dim3(SEQ / 128, HEADS, BATCH), 256, 0, stream>>>(qb_, kb_, vb_, ao);

  // 5) proj GEMM + residual(x) -> x1 (fp32)
  gemm_bf16<1><<<dim3(EMBED / BN, MTOK / BM), 256, 0, stream>>>(
      ao, wp, proj_b, x, x1, nullptr, nullptr, nullptr, nullptr, MTOK, EMBED, EMBED);

  // 6) LN2 -> bf16 (reuse xn)
  layernorm_bf16<<<MTOK, 256, 0, stream>>>(x1, ln2_g, ln2_b, xn);

  // 7) FC1 GEMM + GELU -> h (bf16)
  gemm_bf16<2><<<dim3(HIDDEN / BN, MTOK / BM), 256, 0, stream>>>(
      xn, w1, fc1_b, nullptr, nullptr, hbuf, nullptr, nullptr, nullptr, MTOK, HIDDEN, EMBED);

  // 8) FC2 GEMM + residual(x1) -> d_out (fp32)
  gemm_bf16<1><<<dim3(EMBED / BN, MTOK / BM), 256, 0, stream>>>(
      hbuf, w2, fc2_b, x1, (float*)d_out, nullptr, nullptr, nullptr, nullptr,
      MTOK, EMBED, HIDDEN);
}
